// GraphAttentionLayer_41266045780782
// MI455X (gfx1250) — matrix operand order repaired and checked by reading, not compiled, not run on hardware
//
#include <hip/hip_runtime.h>
#include <hip/hip_bf16.h>

#define NN 8192
#define FF 128
#define HH 4
#define LOG2E 1.4426950408889634f

typedef __attribute__((ext_vector_type(16))) __bf16 v16bf;
typedef __attribute__((ext_vector_type(8)))  float  v8f;

union V16U { uint4 q[2]; unsigned w[8]; v16bf v; };

// Pack two f32 -> two bf16 (round-half-up) in ONE v_perm_b32 + 2 adds.
__device__ __forceinline__ unsigned pack_bf16_2(float lo, float hi) {
  const unsigned a = __builtin_bit_cast(unsigned, hi) + 0x8000u;
  const unsigned b = __builtin_bit_cast(unsigned, lo) + 0x8000u;
  return __builtin_amdgcn_perm(a, b, 0x07060302u);  // {a[3],a[2],b[3],b[2]}
}
__device__ __forceinline__ unsigned short f2bf_bits(float f) {
  unsigned u = __builtin_bit_cast(unsigned, f);
  u += 0x7fffu + ((u >> 16) & 1u);
  return (unsigned short)(u >> 16);
}
__device__ __forceinline__ float bf2f(unsigned short b) {
  const unsigned u = ((unsigned)b) << 16;
  return __builtin_bit_cast(float, u);
}
__device__ __forceinline__ v16bf mk16(uint4 lo, uint4 hi) {
  V16U u; u.q[0] = lo; u.q[1] = hi; return u.v;
}

// ---------------------------------------------------------------------------
// Kernel 1: proj[h] = x @ W[h], stored transposed bf16: projT[h][feat][node].
// One block = (head, 16-row tile); 8 waves cover the 128 output features.
// ---------------------------------------------------------------------------
__global__ __launch_bounds__(256) void gat_proj_kernel(
    const float* __restrict__ x, const float* __restrict__ W,
    unsigned short* __restrict__ projT) {
  __shared__ alignas(16) unsigned short WT[FF * FF];   // W[h] transposed, 32 KB
  const int blk = blockIdx.x;
  const int h   = blk >> 9;              // NN/16 = 512 row-tiles per head
  const int i0  = (blk & 511) << 4;
  const int tid = threadIdx.x;

  const float* Wh = W + h * FF * FF;     // [f_in][f_out]
  for (int idx = tid; idx < FF * FF; idx += 256) {
    const int f = idx >> 7, o = idx & 127;
    WT[o * FF + f] = f2bf_bits(Wh[idx]); // transpose -> [o][f]
  }
  __syncthreads();

  const int lane = tid & 31, wid = tid >> 5;
  const int r = lane & 15, hi = lane >> 4;
  const int c0 = wid << 4;
  v8f acc = {0.f, 0.f, 0.f, 0.f, 0.f, 0.f, 0.f, 0.f};

  const float* xr = x + (size_t)(i0 + r) * FF;
#pragma unroll
  for (int k0 = 0; k0 < FF; k0 += 32) {
    // A operand (16-bit 16x32): lane half selects K 0-7/16-23 vs 8-15/24-31
    const float4 f0 = *(const float4*)(xr + k0 + hi * 8);
    const float4 f1 = *(const float4*)(xr + k0 + hi * 8 + 4);
    const float4 f2 = *(const float4*)(xr + k0 + hi * 8 + 16);
    const float4 f3 = *(const float4*)(xr + k0 + hi * 8 + 20);
    V16U au;
    au.w[0] = pack_bf16_2(f0.x, f0.y); au.w[1] = pack_bf16_2(f0.z, f0.w);
    au.w[2] = pack_bf16_2(f1.x, f1.y); au.w[3] = pack_bf16_2(f1.z, f1.w);
    au.w[4] = pack_bf16_2(f2.x, f2.y); au.w[5] = pack_bf16_2(f2.z, f2.w);
    au.w[6] = pack_bf16_2(f3.x, f3.y); au.w[7] = pack_bf16_2(f3.z, f3.w);
    // B operand: lane = output feature, K contiguous in transposed LDS image
    const uint4 b0 = *(const uint4*)&WT[(c0 + r) * FF + k0 + hi * 8];
    const uint4 b1 = *(const uint4*)&WT[(c0 + r) * FF + k0 + hi * 8 + 16];
    acc = __builtin_amdgcn_wmma_f32_16x16x32_bf16(
        false, au.v, false, mk16(b0, b1), (short)0, acc, false, false);
  }
  // C layout: lane = col, VGPR v = row (+8 for upper lane half)
  uint4 st;
  unsigned* sw = (unsigned*)&st;
  sw[0] = pack_bf16_2(acc[0], acc[1]);
  sw[1] = pack_bf16_2(acc[2], acc[3]);
  sw[2] = pack_bf16_2(acc[4], acc[5]);
  sw[3] = pack_bf16_2(acc[6], acc[7]);
  *(uint4*)(projT + (size_t)(h * FF + c0 + r) * NN + i0 + hi * 8) = st;
}

// ---------------------------------------------------------------------------
// Kernel 2: s_src2[h,i] = log2(e) * proj[h,i,:]·a_src[h,:]   (base-2 domain,
// valid because leaky_relu is positively homogeneous). Coalesced over nodes.
// ---------------------------------------------------------------------------
__global__ __launch_bounds__(256) void gat_scores_kernel(
    const unsigned short* __restrict__ projT,
    const float* __restrict__ a_src, const float* __restrict__ a_tgt,
    float* __restrict__ s_src2, float* __restrict__ s_tgt2) {
  const int g = blockIdx.x * 256 + threadIdx.x;   // HH*NN threads
  const int h = g >> 13;
  const int i = g & (NN - 1);
  const unsigned short* base = projT + (size_t)h * FF * NN + i;
  float as = 0.f, at = 0.f;
#pragma unroll 4
  for (int o = 0; o < FF; ++o) {
    const float p = bf2f(base[(size_t)o * NN]);
    as += p * a_src[h * FF + o];
    at += p * a_tgt[h * FF + o];
  }
  s_src2[g] = as * LOG2E;
  s_tgt2[g] = at * LOG2E;
}

__global__ __launch_bounds__(256) void gat_zero_kernel(float* __restrict__ p, int n) {
  const int g = blockIdx.x * 256 + threadIdx.x;
  if (g < n) p[g] = 0.f;
}

// ---------------------------------------------------------------------------
// Kernel 3: colsum[h,j] = sum_i exp(mask + leaky(s_src+s_tgt))  — softmax over
// axis 1 => COLUMN sums. mask is 0 / -1e9, so exp(masked)==0 exactly: replace
// "+m" by an edge select, and skip the whole body when all 32 lanes' columns
// are masked (ballot). s_src chunk staged in LDS as [i][4 heads].
// ---------------------------------------------------------------------------
__global__ __launch_bounds__(256) void gat_colsum_kernel(
    const float* __restrict__ mask, const float* __restrict__ s_src2,
    const float* __restrict__ s_tgt2, float* __restrict__ colsum) {
  __shared__ alignas(16) float ls[512][4];   // 8 KB
  const int jb = blockIdx.x & 31;            // 32 column blocks of 256
  const int ib = blockIdx.x >> 5;            // 16 row chunks of 512
  const int j  = jb * 256 + threadIdx.x;
  const int ibeg = ib * 512;
  for (int idx = threadIdx.x; idx < 2048; idx += 256) {
    const int h = idx >> 9, ii = idx & 511;
    ls[ii][h] = s_src2[h * NN + ibeg + ii];  // coalesced global read
  }
  __syncthreads();

  float st[HH], acc[HH];
#pragma unroll
  for (int h = 0; h < HH; ++h) { st[h] = s_tgt2[h * NN + j]; acc[h] = 0.f; }

  for (int ii = 0; ii < 512; ++ii) {
    const float m = mask[(size_t)(ibeg + ii) * NN + j];  // coalesced
    const bool edge = m > -0.5f;
    if (__builtin_amdgcn_ballot_w32(edge)) {             // uniform skip
      const float4 ssv = *(const float4*)ls[ii];         // LDS broadcast b128
      float t;
      t = ssv.x + st[0]; t = fmaxf(t, 0.2f * t); acc[0] += edge ? exp2f(t) : 0.f;
      t = ssv.y + st[1]; t = fmaxf(t, 0.2f * t); acc[1] += edge ? exp2f(t) : 0.f;
      t = ssv.z + st[2]; t = fmaxf(t, 0.2f * t); acc[2] += edge ? exp2f(t) : 0.f;
      t = ssv.w + st[3]; t = fmaxf(t, 0.2f * t); acc[3] += edge ? exp2f(t) : 0.f;
    }
  }
#pragma unroll
  for (int h = 0; h < HH; ++h) atomicAdd(&colsum[h * NN + j], acc[h]);
}

__global__ __launch_bounds__(256) void gat_loginv_kernel(float* __restrict__ colsum) {
  const int g = blockIdx.x * 256 + threadIdx.x;   // HH*NN threads
  colsum[g] = -log2f(colsum[g]);                  // fold 1/denom into exponent
}

// ---------------------------------------------------------------------------
// Kernel 4 (hot): out[i,:] = mean_h elu( sum_j attn[h,i,j]*proj[h,j,:] + x[i,:] )
// One block per 16-row tile; ALL 4 heads fused => mask streamed exactly once.
// 64-wide j steps, double-buffered LDS attn tile => ONE barrier per 8 WMMAs.
// ---------------------------------------------------------------------------
#define JSTEP 64
__global__ __launch_bounds__(256) void gat_attn_kernel(
    const float* __restrict__ mask, const float* __restrict__ x,
    const unsigned short* __restrict__ projT,
    const float* __restrict__ s_src2, const float* __restrict__ s_tgt2,
    const float* __restrict__ loginv2, float* __restrict__ out) {
  __shared__ alignas(16) unsigned short Atile[2][HH][16][72];  // 18 KB

  const int tid = threadIdx.x;
  const int i0  = blockIdx.x << 4;

  // --- A-tile producer role: 4 cols x 4 heads per thread per step ---
  const int arow = tid >> 4;            // 0..15
  const int ac   = (tid & 15) << 2;     // 0,4,..,60
  float ss2[HH];
#pragma unroll
  for (int h = 0; h < HH; ++h) ss2[h] = s_src2[h * NN + i0 + arow];
  const float* mrow = mask + (size_t)(i0 + arow) * NN;

  // --- WMMA consumer role ---
  const int lane = tid & 31, wid = tid >> 5;
  const int r = lane & 15, hi = lane >> 4;
  const int c0 = wid << 4;
  const v8f vzero = {0.f, 0.f, 0.f, 0.f, 0.f, 0.f, 0.f, 0.f};
  v8f acc[HH];
  const unsigned short* bb[HH];
#pragma unroll
  for (int h = 0; h < HH; ++h) {
    acc[h] = vzero;
    bb[h] = projT + (size_t)(h * FF + c0 + r) * NN;
  }

  for (int s = 0; s < NN / JSTEP; ++s) {
    const int j = s * JSTEP;
    // ---- produce attn tile for this step into buffer (s&1) ----
    unsigned short (*Ab)[16][72] = Atile[s & 1];
    const float4 mv = *(const float4*)(mrow + j + ac);
    __builtin_prefetch(mrow + j + JSTEP + ac, 0, 0);   // global_prefetch_b8
    const bool e0 = mv.x > -0.5f, e1 = mv.y > -0.5f;
    const bool e2 = mv.z > -0.5f, e3 = mv.w > -0.5f;
#pragma unroll
    for (int h = 0; h < HH; ++h) {
      const float4 st = *(const float4*)(s_tgt2 + h * NN + j + ac);
      const float4 li = *(const float4*)(loginv2 + h * NN + j + ac);
      float t0 = ss2[h] + st.x; t0 = fmaxf(t0, 0.2f * t0) + li.x;
      float t1 = ss2[h] + st.y; t1 = fmaxf(t1, 0.2f * t1) + li.y;
      float t2 = ss2[h] + st.z; t2 = fmaxf(t2, 0.2f * t2) + li.z;
      float t3 = ss2[h] + st.w; t3 = fmaxf(t3, 0.2f * t3) + li.w;
      const float a0 = e0 ? exp2f(t0) : 0.f;
      const float a1 = e1 ? exp2f(t1) : 0.f;
      const float a2 = e2 ? exp2f(t2) : 0.f;
      const float a3 = e3 ? exp2f(t3) : 0.f;
      uint2 pk;
      pk.x = pack_bf16_2(a0, a1);
      pk.y = pack_bf16_2(a2, a3);
      *(uint2*)&Ab[h][arow][ac] = pk;
    }
    __syncthreads();   // single barrier/step: double buffering covers WAR
    // ---- consume: 2 K-blocks x 4 heads = 8 WMMAs ----
#pragma unroll
    for (int kk = 0; kk < 2; ++kk) {
#pragma unroll
      for (int h = 0; h < HH; ++h) {
        const uint4 a0 = *(const uint4*)&Ab[h][r][kk * 32 + hi * 8];
        const uint4 a1 = *(const uint4*)&Ab[h][r][kk * 32 + 16 + hi * 8];
        const uint4 b0 = *(const uint4*)(bb[h] + j + kk * 32 + hi * 8);
        const uint4 b1 = *(const uint4*)(bb[h] + j + kk * 32 + 16 + hi * 8);
        acc[h] = __builtin_amdgcn_wmma_f32_16x16x32_bf16(
            false, mk16(a0, a1), false, mk16(b0, b1), (short)0, acc[h],
            false, false);
      }
    }
  }

  // Epilogue: residual + ELU + head mean, direct store (heads fused in-block).
#pragma unroll
  for (int v = 0; v < 8; ++v) {
    const int i   = i0 + hi * 8 + v;
    const int col = c0 + r;
    const float resid = x[i * FF + col];
    float sum = 0.f;
#pragma unroll
    for (int h = 0; h < HH; ++h) {
      float t = acc[h][v] + resid;
      t = (t > 0.f) ? t : (__expf(t) - 1.f);   // elu
      sum += t;
    }
    out[i * FF + col] = 0.25f * sum;
  }
}

// ---------------------------------------------------------------------------
extern "C" void kernel_launch(void* const* d_in, const int* in_sizes, int n_in,
                              void* d_out, int out_size, void* d_ws,
                              size_t ws_size, hipStream_t stream) {
  const float* x     = (const float*)d_in[0];   // [8192,128]
  const float* mask  = (const float*)d_in[1];   // [8192,8192]
  const float* W     = (const float*)d_in[2];   // [4,128,128]
  const float* a_src = (const float*)d_in[3];   // [4,128]
  const float* a_tgt = (const float*)d_in[4];   // [4,128]
  float* out = (float*)d_out;

  char* ws = (char*)d_ws;
  unsigned short* projT = (unsigned short*)ws;             // 8 MiB bf16
  float* s_src2 = (float*)(ws + (size_t)HH * FF * NN * 2); // 128 KB
  float* s_tgt2 = s_src2 + HH * NN;                        // 128 KB
  float* colsum = s_tgt2 + HH * NN;                        // 128 KB (-> -log2)

  gat_zero_kernel  <<<(HH * NN) / 256, 256, 0, stream>>>(colsum, HH * NN);
  gat_proj_kernel  <<<HH * (NN / 16), 256, 0, stream>>>(x, W, projT);
  gat_scores_kernel<<<(HH * NN) / 256, 256, 0, stream>>>(projT, a_src, a_tgt,
                                                         s_src2, s_tgt2);
  gat_colsum_kernel<<<512, 256, 0, stream>>>(mask, s_src2, s_tgt2, colsum);
  gat_loginv_kernel<<<(HH * NN) / 256, 256, 0, stream>>>(colsum);
  gat_attn_kernel  <<<NN / 16, 256, 0, stream>>>(mask, x, projT, s_src2,
                                                 s_tgt2, colsum, out);
}
